// CAPMemory_45329084842481
// MI455X (gfx1250) — compile-verified, matrix operand order checked by reading, not run on hardware
//
#include <hip/hip_runtime.h>
#include <hip/hip_bf16.h>
#include <math.h>

// Problem constants (match reference)
#define B_SZ   256
#define D_SZ   2048
#define NCAMS  6
#define CCLS   2000
#define NTOT   (NCAMS * CCLS)   // 12000
#define BETAI  20.0f            // 1/BETA, BETA = 0.05
#define BG_KNN 50
#define MASK_VAL -10000.0f

// GEMM tiling
#define BN  32                  // N columns per workgroup (12000 / 32 = 375 blocks)
#define BK  32                  // K tile (one bf16 WMMA K-depth)
#define LDS_STRIDE 40           // padded row stride in bf16 (80B, 16B-aligned, spreads banks)

typedef __bf16 bf16;
typedef __attribute__((ext_vector_type(4)))  bf16  v4bf;
typedef __attribute__((ext_vector_type(8)))  bf16  v8bf;
typedef __attribute__((ext_vector_type(16))) bf16  v16bf;
typedef __attribute__((ext_vector_type(8)))  float v8f;

// ---------------------------------------------------------------------------
// Kernel 1: sims[256 x 12000] = feats[256 x 2048] * mem[12000 x 2048]^T
// bf16 inputs (converted on the fly), f32 accumulate via v_wmma_f32_16x16x32_bf16.
// Each block: full M=256, one 32-wide N stripe -> memory bank streamed once.
// ---------------------------------------------------------------------------
__global__ __launch_bounds__(256)
void capmem_gemm_bf16(const float* __restrict__ feats,
                      const float* __restrict__ mem,
                      float* __restrict__ sims)
{
    __shared__ bf16 ldsA[B_SZ * LDS_STRIDE];   // 256 x (32+pad) bf16 = 20 KB
    __shared__ bf16 ldsB[BN   * LDS_STRIDE];   //  32 x (32+pad) bf16 = 2.5 KB

    const int tid  = threadIdx.x;
    const int lane = tid & 31;
    const int wave = tid >> 5;                 // 0..7
    const int n0   = blockIdx.x * BN;

    const int ntile = wave & 1;                // 0..1 (16-col sub-tile)
    const int mgrp  = wave >> 1;               // 0..3 (rows mgrp*64 .. +63)

    // ISA fragment layout: lanes 0-15 hold K {0..7,16..23}, lanes 16-31 hold K {8..15,24..31}
    const int half = lane >> 4;
    const int l15  = lane & 15;
    const int kc0  = half * 8;
    const int kc1  = 16 + half * 8;

    v8f acc0 = {0}, acc1 = {0}, acc2 = {0}, acc3 = {0};

    const int bcol = tid >> 3;                 // B loader: col 0..31
    const int bk   = (tid & 7) * 4;            // B loader: k offset 0..28

    for (int k0 = 0; k0 < D_SZ; k0 += BK) {
        // Prefetch next B tile (emits global_prefetch_b8)
        if (k0 + BK < D_SZ)
            __builtin_prefetch(mem + (size_t)(n0 + bcol) * D_SZ + (k0 + BK) + bk, 0, 1);

        // --- stage A tile: thread t converts row t, k0..k0+31 ---
        {
            const float* g = feats + (size_t)tid * D_SZ + k0;
            #pragma unroll
            for (int c = 0; c < 4; ++c) {
                float4 u = *(const float4*)(g + c * 8);
                float4 v = *(const float4*)(g + c * 8 + 4);
                v8bf o = { (bf16)u.x, (bf16)u.y, (bf16)u.z, (bf16)u.w,
                           (bf16)v.x, (bf16)v.y, (bf16)v.z, (bf16)v.w };
                *(v8bf*)&ldsA[tid * LDS_STRIDE + c * 8] = o;
            }
        }
        // --- stage B tile: 32 cols x 32 K ---
        {
            float4 u = *(const float4*)(mem + (size_t)(n0 + bcol) * D_SZ + k0 + bk);
            v4bf o = { (bf16)u.x, (bf16)u.y, (bf16)u.z, (bf16)u.w };
            *(v4bf*)&ldsB[bcol * LDS_STRIDE + bk] = o;
        }
        __syncthreads();

        // --- B fragment (lane owns one N column) ---
        const bf16* bp = &ldsB[(ntile * 16 + l15) * LDS_STRIDE];
        v8bf blo = *(const v8bf*)(bp + kc0);
        v8bf bhi = *(const v8bf*)(bp + kc1);
        v16bf bfrag = __builtin_shufflevector(blo, bhi,
            0,1,2,3,4,5,6,7,8,9,10,11,12,13,14,15);

        // --- 4 M sub-tiles x WMMA ---
        #pragma unroll
        for (int i = 0; i < 4; ++i) {
            const bf16* ap = &ldsA[(mgrp * 64 + i * 16 + l15) * LDS_STRIDE];
            v8bf alo = *(const v8bf*)(ap + kc0);
            v8bf ahi = *(const v8bf*)(ap + kc1);
            v16bf afrag = __builtin_shufflevector(alo, ahi,
                0,1,2,3,4,5,6,7,8,9,10,11,12,13,14,15);
            v8f c = (i == 0) ? acc0 : (i == 1) ? acc1 : (i == 2) ? acc2 : acc3;
            c = __builtin_amdgcn_wmma_f32_16x16x32_bf16(
                    false, afrag, false, bfrag, (short)0, c, false, false);
            if (i == 0) acc0 = c; else if (i == 1) acc1 = c;
            else if (i == 2) acc2 = c; else acc3 = c;
        }
        __syncthreads();
    }

    // --- write out: C/D layout: VGPR r -> M = r (lanes<16) / r+8 (lanes>=16) ---
    const int ncol = n0 + ntile * 16 + l15;
    #pragma unroll
    for (int i = 0; i < 4; ++i) {
        v8f c = (i == 0) ? acc0 : (i == 1) ? acc1 : (i == 2) ? acc2 : acc3;
        const int rbase = mgrp * 64 + i * 16 + half * 8;
        #pragma unroll
        for (int r = 0; r < 8; ++r)
            sims[(size_t)(rbase + r) * NTOT + ncol] = c[r];
    }
}

// ---------------------------------------------------------------------------
// Kernel 2: per-row loss. One block per row; row sims cached in LDS.
// ---------------------------------------------------------------------------
__global__ __launch_bounds__(256)
void capmem_rowloss(const float* __restrict__ sims,
                    const int*   __restrict__ cams,
                    const int*   __restrict__ labels,
                    float*       __restrict__ accums)  // [0..5]=ce_sum [6..11]=assoc_sum [12..17]=cnt
{
    __shared__ float srow[NTOT];         // 48 KB
    __shared__ float red [256];
    __shared__ float lmax[256];  __shared__ int lidx[256];
    __shared__ float rmax[256];  __shared__ int ridx[256];
    __shared__ float posv[NCAMS];
    __shared__ float negv[BG_KNN];
    __shared__ int   widx;

    const int row = blockIdx.x;
    const int tid = threadIdx.x;

    for (int j = tid; j < NTOT; j += 256)
        srow[j] = sims[(size_t)row * NTOT + j];
    __syncthreads();

    const int cam   = cams[row];
    const int label = labels[row];
    const int base  = cam * CCLS;

    // ---- per-camera cross entropy: lse over own-camera 2000 block ----
    float m = -1e30f;
    for (int j = tid; j < CCLS; j += 256) m = fmaxf(m, srow[base + j]);
    red[tid] = m; __syncthreads();
    for (int s = 128; s > 0; s >>= 1) {
        if (tid < s) red[tid] = fmaxf(red[tid], red[tid + s]);
        __syncthreads();
    }
    const float M = red[0];
    __syncthreads();

    float psum = 0.0f;
    for (int j = tid; j < CCLS; j += 256)
        psum += __expf((srow[base + j] - M) * BETAI);
    red[tid] = psum; __syncthreads();
    for (int s = 128; s > 0; s >>= 1) {
        if (tid < s) red[tid] += red[tid + s];
        __syncthreads();
    }
    const float lse = __logf(red[0]) + M * BETAI;
    const float ce  = lse - srow[base + label] * BETAI;
    __syncthreads();

    // ---- mask the 6 positives (one per camera) ----
    if (tid == 0) {
        for (int j = 0; j < NCAMS; ++j) {
            const int idx = j * CCLS + label;
            posv[j]   = srow[idx];
            srow[idx] = MASK_VAL;
        }
    }
    __syncthreads();

    // ---- top-50 hard negatives: cached per-thread local maxima ----
    float lm = -1e30f; int li = tid;
    for (int j = tid; j < NTOT; j += 256) {
        const float v = srow[j];
        if (v > lm) { lm = v; li = j; }
    }
    lmax[tid] = lm; lidx[tid] = li;
    __syncthreads();

    for (int it = 0; it < BG_KNN; ++it) {
        rmax[tid] = lmax[tid]; ridx[tid] = lidx[tid];
        __syncthreads();
        for (int s = 128; s > 0; s >>= 1) {
            if (tid < s && rmax[tid + s] > rmax[tid]) {
                rmax[tid] = rmax[tid + s]; ridx[tid] = ridx[tid + s];
            }
            __syncthreads();
        }
        if (tid == 0) { negv[it] = rmax[0]; widx = ridx[0]; }
        __syncthreads();
        const int w = widx;
        if (tid == (w & 255)) {            // owner thread (strided by 256) rescans
            srow[w] = MASK_VAL;
            float nm = -1e30f; int ni = tid;
            for (int j = tid; j < NTOT; j += 256) {
                const float v = srow[j];
                if (v > nm) { nm = v; ni = j; }
            }
            lmax[tid] = nm; lidx[tid] = ni;
        }
        __syncthreads();
    }

    // ---- associate loss over 6 pos + 50 neg logits ----
    if (tid == 0) {
        float m2 = -1e30f;
        for (int j = 0; j < NCAMS;  ++j) m2 = fmaxf(m2, posv[j]);
        for (int j = 0; j < BG_KNN; ++j) m2 = fmaxf(m2, negv[j]);
        float s2 = 0.0f;
        for (int j = 0; j < NCAMS;  ++j) s2 += __expf((posv[j] - m2) * BETAI);
        for (int j = 0; j < BG_KNN; ++j) s2 += __expf((negv[j] - m2) * BETAI);
        const float lse2 = __logf(s2) + m2 * BETAI;
        float assoc = 0.0f;
        for (int j = 0; j < NCAMS; ++j) assoc += (lse2 - posv[j] * BETAI);
        assoc *= (1.0f / NCAMS);

        atomicAdd(&accums[cam],             ce);
        atomicAdd(&accums[NCAMS + cam],     assoc);
        atomicAdd(&accums[2 * NCAMS + cam], 1.0f);
    }
}

// ---------------------------------------------------------------------------
// Init / finalize
// ---------------------------------------------------------------------------
__global__ void capmem_init(float* accums) {
    if (threadIdx.x < 3 * NCAMS) accums[threadIdx.x] = 0.0f;
}

__global__ void capmem_final(const float* __restrict__ accums, float* __restrict__ out) {
    if (threadIdx.x == 0) {
        float loss = 0.0f;
        for (int c = 0; c < NCAMS; ++c) {
            const float cnt = accums[2 * NCAMS + c];
            if (cnt > 0.0f)
                loss += (accums[c] + 0.5f * accums[NCAMS + c]) / cnt;
        }
        out[0] = loss;
    }
}

// ---------------------------------------------------------------------------
// Launch: init -> GEMM -> per-row loss -> finalize (all on `stream`)
// Workspace: sims 256*12000 f32 (~12.3 MB) + 18 f32 accumulators.
// ---------------------------------------------------------------------------
extern "C" void kernel_launch(void* const* d_in, const int* in_sizes, int n_in,
                              void* d_out, int out_size, void* d_ws, size_t ws_size,
                              hipStream_t stream) {
    const float* feats  = (const float*)d_in[0];   // [256, 2048]
    const float* mem    = (const float*)d_in[1];   // [6, 2000, 2048]
    const int*   cams   = (const int*)d_in[2];     // [256]
    const int*   labels = (const int*)d_in[3];     // [256]

    float* sims  = (float*)d_ws;
    float* accum = (float*)((char*)d_ws + (size_t)B_SZ * NTOT * sizeof(float));

    capmem_init     <<<1, 64, 0, stream>>>(accum);
    capmem_gemm_bf16<<<NTOT / BN, 256, 0, stream>>>(feats, mem, sims);
    capmem_rowloss  <<<B_SZ, 256, 0, stream>>>(sims, cams, labels, accum);
    capmem_final    <<<1, 32, 0, stream>>>(accum, (float*)d_out);
}
